// LinearNO_block_2765958939223
// MI455X (gfx1250) — matrix operand order repaired and checked by reading, not compiled, not run on hardware
//
#include <hip/hip_runtime.h>
#include <hip/hip_bf16.h>
#include <math.h>

// Problem dims (fixed by the reference)
#define BB   8
#define NN   8192
#define CC   256
#define HH   8
#define DD   32
#define SS   32
#define HIDD 1024
#define NTOK (BB * NN)   // 65536 tokens

typedef __attribute__((ext_vector_type(16))) __bf16 bf16x16;
typedef __attribute__((ext_vector_type(8)))  __bf16 bf16x8;
typedef __attribute__((ext_vector_type(8)))  float  f32x8;
typedef __attribute__((ext_vector_type(4)))  unsigned int u32x4;
typedef __attribute__((ext_vector_type(8)))  int i32x8;
typedef __attribute__((ext_vector_type(4)))  int i32x4;
typedef __hip_bfloat16 bf16_t;

#if __has_builtin(__builtin_amdgcn_tensor_load_to_lds) && \
    __has_builtin(__builtin_amdgcn_s_wait_tensorcnt)
#define HAVE_TDM 1
#else
#define HAVE_TDM 0
#endif

// B panel in LDS: 64 cols x 32 K-elems, padded to 40-elem (80B) row stride so
// the 16-lane b128 reads are bank-conflict free (stride 20 banks, order 16).
#define PANEL_STRIDE 40
#define PANEL_ELEMS  (64 * PANEL_STRIDE)

// ---------------------------------------------------------------------------
// helpers
// ---------------------------------------------------------------------------
__device__ inline float wave_sum(float v) {
  #pragma unroll
  for (int o = 16; o > 0; o >>= 1) v += __shfl_xor(v, o, 32);
  return v;
}
__device__ inline float wave_max(float v) {
  #pragma unroll
  for (int o = 16; o > 0; o >>= 1) v = fmaxf(v, __shfl_xor(v, o, 32));
  return v;
}
__device__ inline void atomicMaxF(float* addr, float val) {
  unsigned int* ua = (unsigned int*)addr;
  unsigned int old = *ua;
  while (true) {
    float f = __uint_as_float(old);
    if (f >= val) break;
    unsigned int assumed = old;
    old = atomicCAS(ua, assumed, __float_as_uint(val));
    if (old == assumed) break;
  }
}

// A fragment, 16x32 bf16 (contiguous K order, same as B):
// lanes 0-15 : row = lane,    K = {k..k+15}
// lanes 16-31: row = lane-16, K = {k+16..k+31}
__device__ inline bf16x16 load_a_frag(const bf16_t* A, int lda, int row, int k, int half) {
  const bf16_t* p = A + (size_t)row * lda + k + half * 16;
  union { bf16x16 v; bf16x8 h[2]; } u;
  u.h[0] = *(const bf16x8*)(p);
  u.h[1] = *(const bf16x8*)(p + 8);
  return u.v;
}
// B fragment from the LDS panel: column (lane&15) of the 16-col group, K-half
// per lane-half. p points at panel[col*PANEL_STRIDE + half*16].
__device__ inline bf16x16 load_b_lds(const unsigned short* p) {
  union { bf16x16 v; bf16x8 h[2]; } u;
  u.h[0] = *(const bf16x8*)(p);
  u.h[1] = *(const bf16x8*)(p + 8);
  return u.v;
}

#if HAVE_TDM
// TDM: DMA one 64x32 bf16 W panel (rows = output cols, contiguous along K)
// into LDS with 16B padding after every 64B row (pad_interval=3, pad_amount=3)
// giving the 80B padded row stride the compute expects.
__device__ inline void tdm_load_panel(const bf16_t* W, int K, int colBase, int k,
                                      unsigned lds_off) {
  unsigned long long ga = (unsigned long long)(const void*)(W + (size_t)colBase * K + k);
  u32x4 g0;
  g0[0] = 1u;                                     // count=1, user descriptor
  g0[1] = lds_off;                                // lds_addr (bytes)
  g0[2] = (unsigned)(ga & 0xFFFFFFFFu);           // global_addr[31:0]
  g0[3] = ((unsigned)(ga >> 32) & 0x01FFFFFFu)    // global_addr[56:32]
          | (2u << 30);                           // type = 2 (image/tensor)
  i32x8 g1;
  g1[0] = (1 << 16)                               // data_size = 2 bytes
        | (1 << 20)                               // pad_enable
        | (3 << 22)                               // pad_interval: 16 DW = 64B
        | (3 << 25);                              // pad_amount: 4 DW = 16B
  g1[1] = (int)(32u << 16);                       // tensor_dim0 = 32 (lo16)
  g1[2] = (int)(64u << 16);                       // tensor_dim1 = 64 (lo16)
  g1[3] = (int)(32u << 16);                       // tile_dim0 = 32
  g1[4] = 64;                                     // tile_dim1 = 64, tile_dim2 = 0
  g1[5] = K;                                      // tensor_dim0_stride[31:0]
  g1[6] = 0;
  g1[7] = 0;
  i32x4 z4 = {0, 0, 0, 0};
  i32x8 z8 = {0, 0, 0, 0, 0, 0, 0, 0};
  // 6-arg toolchain form: (g0, g1, g2, g3, g4, cpol)
  __builtin_amdgcn_tensor_load_to_lds(g0, g1, z4, z4, z8, 0);
}
#endif

// ---------------------------------------------------------------------------
// Fused GEMM: out[M,N] = A[M,K](bf16) @ W[N,K]^T(bf16) + bias (+GELU) (+resid)
// block = 256 threads = 8 waves; wave tile = 16x64; block tile = 128x64.
// B panel staged in LDS (TDM async DMA, double-buffered; cooperative fallback).
// ---------------------------------------------------------------------------
template <bool GELU, bool OUT_BF16, bool RES>
__global__ __launch_bounds__(256) void gemm_bf16_wmma(
    const bf16_t* __restrict__ A, const bf16_t* __restrict__ W,
    const float* __restrict__ bias, const float* __restrict__ resid,
    void* __restrict__ out, int M, int N, int K) {
  __shared__ __align__(16) unsigned short bpanel[2][PANEL_ELEMS];
  const int lane  = threadIdx.x & 31;
  const int wave  = threadIdx.x >> 5;
  const int half  = lane >> 4;
  const int lan16 = lane & 15;
  const int rowBase = blockIdx.y * 128 + wave * 16;
  const int colBase = blockIdx.x * 64;
  const int rA = rowBase + lan16;
  const int nsteps = K >> 5;

#if HAVE_TDM
  if (wave == 0)
    tdm_load_panel(W, K, colBase, 0, (unsigned)(unsigned long long)(&bpanel[0][0]));
#endif

  f32x8 acc[4] = {};
  for (int i = 0; i < nsteps; ++i) {
    const int k = i << 5;
#if HAVE_TDM
    if (wave == 0) __builtin_amdgcn_s_wait_tensorcnt(0);   // buf[i&1] DMA done
    __syncthreads();                                       // ..visible to all
    if (wave == 0 && i + 1 < nsteps)
      tdm_load_panel(W, K, colBase, k + 32,
                     (unsigned)(unsigned long long)(&bpanel[(i + 1) & 1][0]));
#else
    // cooperative staging fallback: 256 threads x 8 bf16 each
    __syncthreads();
    {
      const int col = threadIdx.x >> 2;
      const int kk  = (threadIdx.x & 3) * 8;
      *(bf16x8*)&bpanel[0][col * PANEL_STRIDE + kk] =
          *(const bf16x8*)(W + (size_t)(colBase + col) * K + k + kk);
    }
    __syncthreads();
#endif
    const unsigned short* panel = &bpanel[HAVE_TDM ? (i & 1) : 0][0];
    bf16x16 a = load_a_frag(A, K, rA, k, half);
    if (k + 32 < K)
      __builtin_prefetch(A + (size_t)rA * K + k + 32, 0, 1);  // global_prefetch_b8
    // issue all 8 ds_load_b128 first, then 4 back-to-back WMMAs (single
    // s_wait_dscnt; no D->A/B RAW hazard between the WMMAs)
    bf16x16 bfr[4];
    #pragma unroll
    for (int j = 0; j < 4; ++j)
      bfr[j] = load_b_lds(panel + (j * 16 + lan16) * PANEL_STRIDE + half * 16);
    #pragma unroll
    for (int j = 0; j < 4; ++j)
      acc[j] = __builtin_amdgcn_wmma_f32_16x16x32_bf16(
          false, a, false, bfr[j], (short)0, acc[j], false, false);
  }

  // C/D layout: lanes 0-15 -> rows rowBase+i, lanes 16-31 -> rows rowBase+8+i
  const int rOut = rowBase + half * 8;
  #pragma unroll
  for (int j = 0; j < 4; ++j) {
    const int col = colBase + j * 16 + lan16;
    const float bcol = bias[col];
    #pragma unroll
    for (int i = 0; i < 8; ++i) {
      const size_t idx = (size_t)(rOut + i) * N + col;
      float v = acc[j][i] + bcol;
      if (GELU) v = 0.5f * v * (1.0f + erff(v * 0.70710678118654752f));
      if (RES)  v += resid[idx];
      if (OUT_BF16) ((bf16_t*)out)[idx] = __float2bfloat16(v);
      else          ((float*)out)[idx]  = v;
    }
  }
}

// ---------------------------------------------------------------------------
// LayerNorm over C=256, one wave per token, bf16 output
// ---------------------------------------------------------------------------
__global__ __launch_bounds__(256) void ln_kernel(
    const float* __restrict__ x, const float* __restrict__ w,
    const float* __restrict__ bvec, bf16_t* __restrict__ y) {
  const int lane = threadIdx.x & 31;
  const int tok  = blockIdx.x * 8 + (threadIdx.x >> 5);
  const float* row = x + (size_t)tok * CC;
  float v[8];
  float s = 0.f;
  #pragma unroll
  for (int i = 0; i < 8; ++i) { v[i] = row[lane + 32 * i]; s += v[i]; }
  const float mu = wave_sum(s) * (1.0f / CC);
  float q = 0.f;
  #pragma unroll
  for (int i = 0; i < 8; ++i) { float d = v[i] - mu; q += d * d; }
  const float rstd = rsqrtf(wave_sum(q) * (1.0f / CC) + 1e-5f);
  bf16_t* orow = y + (size_t)tok * CC;
  #pragma unroll
  for (int i = 0; i < 8; ++i) {
    const int c = lane + 32 * i;
    orow[c] = __float2bfloat16((v[i] - mu) * rstd * w[c] + bvec[c]);
  }
}

// ---------------------------------------------------------------------------
// Per-head linear attention, stats pipeline (k-softmax over the TOKEN axis)
// ---------------------------------------------------------------------------
__global__ void init_stats(float* Mcol, float* Z, float* Avs) {
  const int i = blockIdx.x * blockDim.x + threadIdx.x;
  if (i < BB * HH * SS) { Mcol[i] = -3.0e38f; Z[i] = 0.f; }
  if (i < BB * HH * SS * DD) Avs[i] = 0.f;
}

// pass 1: column max of k-logits over tokens
__global__ __launch_bounds__(256) void colmax_kernel(
    const float* __restrict__ xmid, const float* __restrict__ Wk,
    float* __restrict__ Mcol) {
  const int lane = threadIdx.x & 31;
  const int wave = threadIdx.x >> 5;
  const int bh = blockIdx.x, b = bh >> 3, h = bh & 7;
  const int n0 = blockIdx.y * 256 + wave * 32;
  float wkrow[32];
  #pragma unroll
  for (int d = 0; d < 32; ++d) wkrow[d] = Wk[lane * 32 + d];
  float mymax = -3.0e38f;
  for (int it = 0; it < 32; ++it) {
    const float xh = xmid[((size_t)(b * NN + n0 + it)) * CC + h * DD + lane];
    float kl = 0.f;
    #pragma unroll
    for (int d = 0; d < 32; ++d) kl += __shfl(xh, d, 32) * wkrow[d];
    mymax = fmaxf(mymax, kl);
  }
  atomicMaxF(&Mcol[bh * SS + lane], mymax);
}

// pass 2: Z[s] = sum_n exp(kl-M), Avs[s,d] = sum_n exp(kl-M)*v[d]
__global__ __launch_bounds__(256) void kvacc_kernel(
    const float* __restrict__ xmid, const float* __restrict__ Wk,
    const float* __restrict__ Wv, const float* __restrict__ Mcol,
    float* __restrict__ Z, float* __restrict__ Avs) {
  __shared__ float kvp[SS * DD];
  __shared__ float zp[SS];
  const int lane = threadIdx.x & 31;
  const int wave = threadIdx.x >> 5;
  for (int i = threadIdx.x; i < SS * DD; i += 256) kvp[i] = 0.f;
  if (threadIdx.x < SS) zp[threadIdx.x] = 0.f;
  __syncthreads();
  const int bh = blockIdx.x, b = bh >> 3, h = bh & 7;
  const int n0 = blockIdx.y * 256 + wave * 32;
  float wkrow[32], wvrow[32];
  #pragma unroll
  for (int d = 0; d < 32; ++d) {
    wkrow[d] = Wk[lane * 32 + d];
    wvrow[d] = Wv[lane * 32 + d];
  }
  const float mcol = Mcol[bh * SS + lane];
  for (int it = 0; it < 32; ++it) {
    const float xh = xmid[((size_t)(b * NN + n0 + it)) * CC + h * DD + lane];
    float kl = 0.f, vv = 0.f;
    #pragma unroll
    for (int d = 0; d < 32; ++d) {
      const float xd = __shfl(xh, d, 32);
      kl += xd * wkrow[d];
      vv += xd * wvrow[d];
    }
    const float w = __expf(kl - mcol);
    atomicAdd(&zp[lane], w);                         // ds_add_f32
    #pragma unroll
    for (int s = 0; s < 32; ++s) {
      const float ws = __shfl(w, s, 32);
      atomicAdd(&kvp[s * DD + lane], ws * vv);       // ds_add_f32
    }
  }
  __syncthreads();
  for (int i = threadIdx.x; i < SS * DD; i += 256) atomicAdd(&Avs[bh * SS * DD + i], kvp[i]);
  if (threadIdx.x < SS) atomicAdd(&Z[bh * SS + threadIdx.x], zp[threadIdx.x]);
}

__global__ void kvnorm_kernel(const float* __restrict__ Avs,
                              const float* __restrict__ Z, float* __restrict__ kv) {
  const int i = blockIdx.x * blockDim.x + threadIdx.x;
  if (i >= BB * HH * SS * DD) return;
  const int s = (i >> 5) & 31, bh = i >> 10;
  kv[i] = Avs[i] / Z[bh * SS + s];
}

// pass 3: q-softmax (over S, within-wave) and qkv = q @ kv, bf16 out
__global__ __launch_bounds__(256) void qkv_kernel(
    const float* __restrict__ xmid, const float* __restrict__ Wq,
    const float* __restrict__ kv, bf16_t* __restrict__ qkvout) {
  __shared__ float kvl[SS * DD];
  const int lane = threadIdx.x & 31;
  const int wave = threadIdx.x >> 5;
  const int bh = blockIdx.x, b = bh >> 3, h = bh & 7;
  for (int i = threadIdx.x; i < SS * DD; i += 256) kvl[i] = kv[bh * SS * DD + i];
  __syncthreads();
  float wqrow[32];
  #pragma unroll
  for (int d = 0; d < 32; ++d) wqrow[d] = Wq[lane * 32 + d];
  const int n0 = blockIdx.y * 256 + wave * 32;
  for (int it = 0; it < 32; ++it) {
    const size_t base = ((size_t)(b * NN + n0 + it)) * CC + h * DD;
    const float xh = xmid[base + lane];
    float ql = 0.f;
    #pragma unroll
    for (int d = 0; d < 32; ++d) ql += __shfl(xh, d, 32) * wqrow[d];
    const float m  = wave_max(ql);
    const float e  = __expf(ql - m);
    const float zs = wave_sum(e);
    const float q  = e / zs;
    float o = 0.f;
    #pragma unroll
    for (int s = 0; s < 32; ++s) o += __shfl(q, s, 32) * kvl[s * DD + lane];
    qkvout[base + lane] = __float2bfloat16(o);
  }
}

__global__ void cvt_kernel(const float* __restrict__ src, bf16_t* __restrict__ dst, int n) {
  const int i = blockIdx.x * blockDim.x + threadIdx.x;
  if (i < n) dst[i] = __float2bfloat16(src[i]);
}

// ---------------------------------------------------------------------------
// launch
// ---------------------------------------------------------------------------
extern "C" void kernel_launch(void* const* d_in, const int* in_sizes, int n_in,
                              void* d_out, int out_size, void* d_ws, size_t ws_size,
                              hipStream_t stream) {
  (void)in_sizes; (void)n_in; (void)out_size; (void)ws_size;
  const float* fx    = (const float*)d_in[0];
  const float* ln1_w = (const float*)d_in[1];
  const float* ln1_b = (const float*)d_in[2];
  const float* Wx    = (const float*)d_in[3];
  const float* bx    = (const float*)d_in[4];
  const float* Wq    = (const float*)d_in[5];
  const float* Wk    = (const float*)d_in[6];
  const float* Wv    = (const float*)d_in[7];
  const float* Wo    = (const float*)d_in[8];
  const float* bo    = (const float*)d_in[9];
  const float* ln2_w = (const float*)d_in[10];
  const float* ln2_b = (const float*)d_in[11];
  const float* W1    = (const float*)d_in[12];
  const float* b1    = (const float*)d_in[13];
  const float* W2    = (const float*)d_in[14];
  const float* b2    = (const float*)d_in[15];

  // workspace layout (region A is reused: xln|xmid|qkv are all dead before the
  // MLP hidden activation overlays them)
  char* ws = (char*)d_ws;
  bf16_t* xln  = (bf16_t*)(ws + 0);            //  32 MB  LN1 output (bf16)
  float*  xmid = (float*)(ws + 33554432ull);   //  64 MB  x projection (f32)
  bf16_t* qkvb = (bf16_t*)(ws + 100663296ull); //  32 MB  attention out (bf16)
  bf16_t* hid  = (bf16_t*)(ws + 0);            // 128 MB  MLP hidden, overlays region A
  float*  hbuf = (float*)(ws + 134217728ull);  //  64 MB  residual stream h (f32)
  bf16_t* yln  = (bf16_t*)(ws + 201326592ull); //  32 MB  LN2 output (bf16)
  char* wb = ws + 234881024ull;                // bf16 weight copies
  bf16_t* Wxb = (bf16_t*)(wb);
  bf16_t* Wob = (bf16_t*)(wb + 131072);
  bf16_t* W1b = (bf16_t*)(wb + 262144);
  bf16_t* W2b = (bf16_t*)(wb + 786432);
  char* st = wb + 1310720;                     // tiny stats
  float* Mcol = (float*)(st);                  // [B*H,S] col max
  float* Zb   = (float*)(st + 8192);           // [B*H,S] col sumexp
  float* Avs  = (float*)(st + 16384);          // [B*H,S,D] weighted sums
  float* kvb  = (float*)(st + 16384 + 262144); // [B*H,S,D] kv

  // weight conversion + stat init
  cvt_kernel<<<(CC * CC + 255) / 256, 256, 0, stream>>>(Wx, Wxb, CC * CC);
  cvt_kernel<<<(CC * CC + 255) / 256, 256, 0, stream>>>(Wo, Wob, CC * CC);
  cvt_kernel<<<(HIDD * CC + 255) / 256, 256, 0, stream>>>(W1, W1b, HIDD * CC);
  cvt_kernel<<<(CC * HIDD + 255) / 256, 256, 0, stream>>>(W2, W2b, CC * HIDD);
  init_stats<<<(BB * HH * SS * DD + 255) / 256, 256, 0, stream>>>(Mcol, Zb, Avs);

  // LN1 -> x projection (WMMA + TDM-staged B panels)
  ln_kernel<<<NTOK / 8, 256, 0, stream>>>(fx, ln1_w, ln1_b, xln);
  gemm_bf16_wmma<false, false, false><<<dim3(CC / 64, NTOK / 128), 256, 0, stream>>>(
      xln, Wxb, bx, nullptr, xmid, NTOK, CC, CC);

  // per-head linear attention (k softmax over token axis -> kv; then q @ kv)
  colmax_kernel<<<dim3(BB * HH, NN / 256), 256, 0, stream>>>(xmid, Wk, Mcol);
  kvacc_kernel<<<dim3(BB * HH, NN / 256), 256, 0, stream>>>(xmid, Wk, Wv, Mcol, Zb, Avs);
  kvnorm_kernel<<<(BB * HH * SS * DD + 255) / 256, 256, 0, stream>>>(Avs, Zb, kvb);
  qkv_kernel<<<dim3(BB * HH, NN / 256), 256, 0, stream>>>(xmid, Wq, kvb, qkvb);

  // output projection + residual(fx) -> h   (WMMA)
  gemm_bf16_wmma<false, false, true><<<dim3(CC / 64, NTOK / 128), 256, 0, stream>>>(
      qkvb, Wob, bo, fx, hbuf, NTOK, CC, CC);

  // LN2 -> MLP1 (+GELU, bf16 out) -> MLP2 (+residual h) -> d_out   (WMMA)
  ln_kernel<<<NTOK / 8, 256, 0, stream>>>(hbuf, ln2_w, ln2_b, yln);
  gemm_bf16_wmma<true, true, false><<<dim3(HIDD / 64, NTOK / 128), 256, 0, stream>>>(
      yln, W1b, b1, nullptr, hid, NTOK, HIDD, CC);
  gemm_bf16_wmma<false, false, true><<<dim3(CC / 64, NTOK / 128), 256, 0, stream>>>(
      hid, W2b, b2, hbuf, (float*)d_out, NTOK, CC, HIDD);
}
